// NemotronHMambaDecoderLayer_18940805776155
// MI455X (gfx1250) — compile-verified, matrix operand order checked by reading, not run on hardware
//
#include <hip/hip_runtime.h>

// ---------------------------------------------------------------------------
// NemotronH Mamba decoder layer for gfx1250 (MI455X, wave32, WMMA + TDM)
// ---------------------------------------------------------------------------

typedef __attribute__((ext_vector_type(16))) __bf16 v16bf;
typedef __attribute__((ext_vector_type(8)))  float  v8f;

#define DEV __device__ __forceinline__

// ---- problem sizes (fixed by the reference) ----
constexpr int BATCH   = 2;
constexpr int LSEQ    = 2048;
constexpr int HS      = 2048;                        // hidden size
constexpr int HEADS   = 64;
constexpr int HDIM    = 64;
constexpr int INTER   = HEADS * HDIM;                // 4096
constexpr int GROUPS  = 8;
constexpr int STATE   = 128;
constexpr int CONV_DIM = INTER + 2 * GROUPS * STATE; // 6144
constexpr int PROJ_N  = INTER + CONV_DIM + HEADS;    // 10304
constexpr int ROWS    = BATCH * LSEQ;                // 4096
constexpr float EPS   = 1e-5f;

// ---- helpers ----
DEV unsigned short f2bf(float f) {
  unsigned u = __float_as_uint(f);
  u += 0x7fffu + ((u >> 16) & 1u);          // round-to-nearest-even
  return (unsigned short)(u >> 16);
}
DEV float silu_f(float x) { return x / (1.0f + __expf(-x)); }

// ===========================================================================
// Tensor Data Mover: 2D tile (rows x 32 bf16) global -> LDS, per ISA ch.8 D#.
// group0: count=1 | lds_addr | global_addr[56:0] | type=2
// group1: data_size=1(2B) | tensor_dim0=K | tensor_dim1=2^20 | tile_dim0=32
//         | tile_dim1=rows | tensor_dim0_stride=ld
// 6-arg builtin form on this toolchain: (v4u, v8i, v4i, v4i, v8i, i32 cpol)
// ===========================================================================
#if __has_builtin(__builtin_amdgcn_tensor_load_to_lds)
#define USE_TDM 1
typedef unsigned tdm_v4u __attribute__((ext_vector_type(4)));
typedef int      tdm_v8i __attribute__((ext_vector_type(8)));
typedef int      tdm_v4i __attribute__((ext_vector_type(4)));

DEV void tdm_load_2d(unsigned lds_off, unsigned long long gaddr,
                     unsigned rows, unsigned kdim, unsigned stride) {
  tdm_v4u g0;
  g0[0] = 1u;                                                   // count=1
  g0[1] = lds_off;                                              // lds_addr
  g0[2] = (unsigned)gaddr;                                      // gaddr[31:0]
  g0[3] = (unsigned)((gaddr >> 32) & 0x01ffffffu) | (2u << 30); // [56:32]|type=2
  tdm_v8i g1;
  g1[0] = (int)(1u << 16);                        // data_size = 1 -> 2 bytes
  g1[1] = (int)((kdim & 0xffffu) << 16);          // tensor_dim0[15:0]
  g1[2] = (int)((kdim >> 16) & 0xffffu);          // tensor_dim0[31:16], dim1 lo=0
  g1[3] = (int)(16u | (32u << 16));               // tensor_dim1 hi (2^20), tile_dim0=32
  g1[4] = (int)rows;                              // tile_dim1 = rows, tile_dim2=0
  g1[5] = (int)stride;                            // tensor_dim0_stride[31:0]
  g1[6] = 0;                                      // stride hi, dim1_stride lo
  g1[7] = 0;
  tdm_v4i z4 = {0, 0, 0, 0};
  tdm_v8i z8 = {0, 0, 0, 0, 0, 0, 0, 0};
  __builtin_amdgcn_tensor_load_to_lds(g0, g1, z4, z4, z8, 0);
}
#else
#define USE_TDM 0
#endif

// ===========================================================================
// 1) weight convert + transpose via LDS tile (coalesced both sides):
//    W (Kd x Nd, f32, row major) -> Wt (Nd x Kd, bf16).  block 32x8.
// ===========================================================================
__global__ __launch_bounds__(256)
void k_convert_wT(const float* __restrict__ W, unsigned short* __restrict__ Wt,
                  int Kd, int Nd) {
  __shared__ float tile[32][33];
  const int tx = threadIdx.x, ty = threadIdx.y;
  const int k0 = blockIdx.x * 32, n0 = blockIdx.y * 32;
#pragma unroll
  for (int i = 0; i < 4; ++i)
    tile[ty + 8 * i][tx] = W[(size_t)(k0 + ty + 8 * i) * Nd + (n0 + tx)];
  __syncthreads();
#pragma unroll
  for (int i = 0; i < 4; ++i)
    Wt[(size_t)(n0 + ty + 8 * i) * Kd + (k0 + tx)] = f2bf(tile[tx][ty + 8 * i]);
}

// ===========================================================================
// 2) residual add + RMSNorm; writes f32 residual (2nd output) + bf16 x_norm
// ===========================================================================
__global__ __launch_bounds__(256)
void k_resid_norm(const float* __restrict__ hs, const float* __restrict__ res,
                  const float* __restrict__ w,
                  float* __restrict__ resid_out,
                  unsigned short* __restrict__ xbf) {
  const int row = blockIdx.x;
  const int tid = threadIdx.x;
  const float* ph = hs  + (size_t)row * HS;
  const float* pr = res + (size_t)row * HS;
  float*       po = resid_out + (size_t)row * HS;
  unsigned short* px = xbf + (size_t)row * HS;

  float v[8];
  float ss = 0.0f;
#pragma unroll
  for (int i = 0; i < 8; ++i) {
    int c = tid + 256 * i;
    float s = ph[c] + pr[c];
    v[i] = s;
    po[c] = s;
    ss += s * s;
  }
  __shared__ float red[256];
  red[tid] = ss;
  __syncthreads();
  for (int s = 128; s > 0; s >>= 1) {
    if (tid < s) red[tid] += red[tid + s];
    __syncthreads();
  }
  const float rs = rsqrtf(red[0] / (float)HS + EPS);
#pragma unroll
  for (int i = 0; i < 8; ++i) {
    int c = tid + 256 * i;
    px[c] = f2bf(v[i] * rs * w[c]);
  }
}

// ===========================================================================
// 3) bf16 WMMA GEMM:  C[M,N](f32) = A[M,K](bf16 row major) x Bt[N,K](bf16)
//    WG = 128 threads = 4 waves (2 M x 2 N); WG tile 128M x 64N.
//    Wave tile 64x32 = 4x2 v_wmma_f32_16x16x32_bf16 accumulators.
//    K staged in 32-wide slabs, double-buffered LDS fed by TDM
//    (tensor_load_to_lds + s_wait_tensorcnt) on the TDM path, or by direct
//    global_load_b128 fragment gathers on the fallback path.
//    Fragment lane layout per CDNA5 ISA 7.12.2: lane = {m|n = lane&15,
//    half = lane>>4}; dword j at kk = 8*half + 2*(j&3) + 16*(j>>2).
// ===========================================================================
__global__ __launch_bounds__(128)
void k_gemm_bf16(const unsigned short* __restrict__ A, int lda,
                 const unsigned short* __restrict__ Bt, int ldb,
                 float* __restrict__ C, int ldc, int Kdim) {
  union Frag { unsigned u[8]; v16bf v; };

  const int tid  = threadIdx.x;
  const int wave = tid >> 5;
  const int lane = tid & 31;
  const int lm   = lane & 15;
  const int lh   = lane >> 4;
  const int wm   = wave & 1;             // wave row (2)
  const int wn   = wave >> 1;            // wave col (2)
  const int mBlk = blockIdx.y * 128;
  const int nBlk = blockIdx.x * 64;
  const int mW   = wm * 64;              // wave offset in block tile
  const int nW   = wn * 32;
  const int koff = lh * 8;

  v8f acc[4][2] = {};

#if USE_TDM
  __shared__ __align__(16) unsigned short sA[2][128 * 32];
  __shared__ __align__(16) unsigned short sB[2][64 * 32];

  auto issue = [&](int cur, int kb) {
    tdm_load_2d((unsigned)(size_t)&sA[cur][0],
                (unsigned long long)(size_t)A + ((size_t)mBlk * lda + kb) * 2,
                128u, (unsigned)Kdim, (unsigned)lda);
    tdm_load_2d((unsigned)(size_t)&sB[cur][0],
                (unsigned long long)(size_t)Bt + ((size_t)nBlk * ldb + kb) * 2,
                64u, (unsigned)Kdim, (unsigned)ldb);
  };

  auto compute = [&](int cur) {
    Frag a[4], b[2];
#pragma unroll
    for (int i = 0; i < 4; ++i) {
      const int r = mW + 16 * i + lm;
#pragma unroll
      for (int j = 0; j < 8; ++j) {
        const int kk = koff + ((j & 3) << 1) + ((j >> 2) << 4);
        a[i].u[j] = *(const unsigned*)&sA[cur][r * 32 + kk];
      }
    }
#pragma unroll
    for (int i = 0; i < 2; ++i) {
      const int r = nW + 16 * i + lm;
#pragma unroll
      for (int j = 0; j < 8; ++j) {
        const int kk = koff + ((j & 3) << 1) + ((j >> 2) << 4);
        b[i].u[j] = *(const unsigned*)&sB[cur][r * 32 + kk];
      }
    }
#pragma unroll
    for (int i = 0; i < 4; ++i)
#pragma unroll
      for (int j = 0; j < 2; ++j)
        acc[i][j] = __builtin_amdgcn_wmma_f32_16x16x32_bf16(
            false, a[i].v, false, b[j].v, (short)0, acc[i][j], false, false);
  };

  if (tid < 32) issue(0, 0);
  int kb = 0;
  for (; kb < Kdim - 32; kb += 32) {
    const int cur = (kb >> 5) & 1;
    if (tid < 32) {
      issue(cur ^ 1, kb + 32);                    // prefetch next slab
      __builtin_amdgcn_s_wait_tensorcnt(2);       // current slab's 2 DMAs done
    }
    __syncthreads();
    compute(cur);
    __syncthreads();                              // done reading before reuse
  }
  {
    const int cur = (kb >> 5) & 1;
    if (tid < 32) __builtin_amdgcn_s_wait_tensorcnt(0);
    __syncthreads();
    compute(cur);
  }
#else
  // fallback: direct global fragment gathers (round-0 style, 4x2 tiles)
  const unsigned short* ar[4];
  const unsigned short* br[2];
#pragma unroll
  for (int i = 0; i < 4; ++i) ar[i] = A  + (size_t)(mBlk + mW + 16 * i + lm) * lda;
#pragma unroll
  for (int i = 0; i < 2; ++i) br[i] = Bt + (size_t)(nBlk + nW + 16 * i + lm) * ldb;
  for (int kb = 0; kb < Kdim; kb += 32) {
    Frag a[4], b[2];
#pragma unroll
    for (int j = 0; j < 8; ++j) {
      const int kk = kb + koff + ((j & 3) << 1) + ((j >> 2) << 4);
#pragma unroll
      for (int i = 0; i < 4; ++i) a[i].u[j] = *(const unsigned*)(ar[i] + kk);
#pragma unroll
      for (int i = 0; i < 2; ++i) b[i].u[j] = *(const unsigned*)(br[i] + kk);
    }
    __builtin_prefetch(ar[0] + kb + 32 + koff, 0, 1);
    __builtin_prefetch(br[0] + kb + 32 + koff, 0, 1);
#pragma unroll
    for (int i = 0; i < 4; ++i)
#pragma unroll
      for (int j = 0; j < 2; ++j)
        acc[i][j] = __builtin_amdgcn_wmma_f32_16x16x32_bf16(
            false, a[i].v, false, b[j].v, (short)0, acc[i][j], false, false);
  }
#endif

  // C/D layout: lane holds col n, vgpr r -> row m = r + 8*lh
#pragma unroll
  for (int i = 0; i < 4; ++i) {
#pragma unroll
    for (int j = 0; j < 2; ++j) {
      const int n = nBlk + nW + 16 * j + lm;
#pragma unroll
      for (int r = 0; r < 8; ++r) {
        const int m = mBlk + mW + 16 * i + r + 8 * lh;
        C[(size_t)m * ldc + n] = acc[i][j][r];
      }
    }
  }
}

// ===========================================================================
// 4) causal depthwise conv (K=4) + bias + SiLU over xBC slice of proj
// ===========================================================================
__global__ __launch_bounds__(256)
void k_conv_silu(const float* __restrict__ proj, const float* __restrict__ cw,
                 const float* __restrict__ cb, float* __restrict__ xconv) {
  long long i = (long long)blockIdx.x * 256 + threadIdx.x;
  if (i >= (long long)ROWS * CONV_DIM) return;
  const int c = (int)(i % CONV_DIM);
  const long long row = i / CONV_DIM;       // b*L + l
  const int l = (int)(row % LSEQ);

  const float w0 = cw[c * 4 + 0], w1 = cw[c * 4 + 1];
  const float w2 = cw[c * 4 + 2], w3 = cw[c * 4 + 3];
  const float* p = proj + row * (long long)PROJ_N + INTER + c;

  float acc = cb[c];
  if (l >= 3) acc = fmaf(p[-3LL * PROJ_N], w0, acc);
  if (l >= 2) acc = fmaf(p[-2LL * PROJ_N], w1, acc);
  if (l >= 1) acc = fmaf(p[-1LL * PROJ_N], w2, acc);
  acc = fmaf(p[0], w3, acc);
  xconv[i] = silu_f(acc);
}

// ===========================================================================
// 5) dt = softplus(dt_raw + bias);  dA = exp(dt * -exp(A_log))
// ===========================================================================
__global__ __launch_bounds__(256)
void k_dt(const float* __restrict__ proj, const float* __restrict__ dt_bias,
          const float* __restrict__ A_log,
          float* __restrict__ dtp, float* __restrict__ dAb) {
  int i = blockIdx.x * 256 + threadIdx.x;     // ROWS*HEADS = 262144
  if (i >= ROWS * HEADS) return;
  const int hd = i & (HEADS - 1);
  const long long row = i >> 6;
  float x = proj[row * (long long)PROJ_N + INTER + CONV_DIM + hd] + dt_bias[hd];
  float sp = (x > 20.0f) ? x : log1pf(__expf(x));
  float a = -__expf(A_log[hd]);
  dtp[i] = sp;
  dAb[i] = __expf(sp * a);
}

// ===========================================================================
// 6) selective scan. One workgroup per (batch, head); 64x128 f32 state in
//    registers (32/thread x 256 threads). B/C staged via LDS each step,
//    N-reduction by wave32 __shfl_xor.
// ===========================================================================
__global__ __launch_bounds__(256)
void k_scan(const float* __restrict__ xconv, const float* __restrict__ dtp,
            const float* __restrict__ dAb, const float* __restrict__ Dp,
            float* __restrict__ y) {
  const int bh   = blockIdx.x;                 // 0..127
  const int b    = bh >> 6;
  const int head = bh & 63;
  const int g    = head >> 3;                  // group = head / (HEADS/GROUPS)
  const int tid  = threadIdx.x;
  const int p    = tid >> 2;
  const int q    = tid & 3;
  const int nb   = q * 32;

  float h[32];
#pragma unroll
  for (int i = 0; i < 32; ++i) h[i] = 0.0f;
  const float Dv = Dp[head];

  __shared__ float sB[STATE];
  __shared__ float sC[STATE];

  for (int l = 0; l < LSEQ; ++l) {
    const long long row = (long long)b * LSEQ + l;
    const float* xr = xconv + row * (long long)CONV_DIM;
    if (tid < 128) sB[tid]       = xr[INTER + g * STATE + tid];
    else           sC[tid - 128] = xr[INTER + GROUPS * STATE + g * STATE + (tid - 128)];
    __syncthreads();

    const float dAv = dAb[row * HEADS + head];
    const float dtv = dtp[row * HEADS + head];
    const float xv  = xr[head * HDIM + p];
    const float dtx = dtv * xv;

    float acc = 0.0f;
#pragma unroll
    for (int i = 0; i < 32; ++i) {
      float hn = fmaf(h[i], dAv, dtx * sB[nb + i]);
      h[i] = hn;
      acc = fmaf(hn, sC[nb + i], acc);
    }
    acc += __shfl_xor(acc, 1, 32);
    acc += __shfl_xor(acc, 2, 32);
    if (q == 0) y[(row * HEADS + head) * HDIM + p] = fmaf(xv, Dv, acc);
    __syncthreads();
  }
}

// ===========================================================================
// 7) gated RMSNorm: yb = bf16( rmsnorm(y * silu(z)) * gw ),  row = 4096 wide
// ===========================================================================
__global__ __launch_bounds__(256)
void k_gate_norm(const float* __restrict__ y, const float* __restrict__ proj,
                 const float* __restrict__ gw, unsigned short* __restrict__ ybf) {
  const int row = blockIdx.x;
  const int tid = threadIdx.x;
  const float* py = y    + (size_t)row * INTER;
  const float* pz = proj + (size_t)row * PROJ_N;   // z = first INTER columns

  float v[16];
  float ss = 0.0f;
#pragma unroll
  for (int i = 0; i < 16; ++i) {
    int c = tid + 256 * i;
    float z = pz[c];
    float t = py[c] * silu_f(z);
    v[i] = t;
    ss += t * t;
  }
  __shared__ float red[256];
  red[tid] = ss;
  __syncthreads();
  for (int s = 128; s > 0; s >>= 1) {
    if (tid < s) red[tid] += red[tid + s];
    __syncthreads();
  }
  const float rs = rsqrtf(red[0] / (float)INTER + EPS);
  unsigned short* po = ybf + (size_t)row * INTER;
#pragma unroll
  for (int i = 0; i < 16; ++i) {
    int c = tid + 256 * i;
    po[c] = f2bf(v[i] * rs * gw[c]);
  }
}

// ===========================================================================
// workspace layout
// ===========================================================================
constexpr size_t alg(size_t x) { return (x + 255) & ~(size_t)255; }
constexpr size_t OFF_XBF  = 0;
constexpr size_t OFF_WIN  = alg(OFF_XBF  + (size_t)ROWS * HS * 2);
constexpr size_t OFF_WOUT = alg(OFF_WIN  + (size_t)PROJ_N * HS * 2);
constexpr size_t OFF_PROJ = alg(OFF_WOUT + (size_t)HS * INTER * 2);
constexpr size_t OFF_XC   = alg(OFF_PROJ + (size_t)ROWS * PROJ_N * 4);
constexpr size_t OFF_DTP  = alg(OFF_XC   + (size_t)ROWS * CONV_DIM * 4);
constexpr size_t OFF_DA   = alg(OFF_DTP  + (size_t)ROWS * HEADS * 4);
constexpr size_t OFF_Y    = alg(OFF_DA   + (size_t)ROWS * HEADS * 4);
constexpr size_t OFF_YBF  = alg(OFF_Y    + (size_t)ROWS * INTER * 4);

extern "C" void kernel_launch(void* const* d_in, const int* in_sizes, int n_in,
                              void* d_out, int out_size, void* d_ws, size_t ws_size,
                              hipStream_t stream) {
  const float* hs       = (const float*)d_in[0];
  const float* res      = (const float*)d_in[1];
  const float* norm_w   = (const float*)d_in[2];
  const float* W_in     = (const float*)d_in[3];
  const float* conv_w   = (const float*)d_in[4];
  const float* conv_b   = (const float*)d_in[5];
  const float* A_log    = (const float*)d_in[6];
  const float* Dp       = (const float*)d_in[7];
  const float* dt_bias  = (const float*)d_in[8];
  const float* gate_w   = (const float*)d_in[9];
  const float* W_out    = (const float*)d_in[10];

  char* ws = (char*)d_ws;
  unsigned short* xbf   = (unsigned short*)(ws + OFF_XBF);
  unsigned short* WtIn  = (unsigned short*)(ws + OFF_WIN);
  unsigned short* WtOut = (unsigned short*)(ws + OFF_WOUT);
  float*          proj  = (float*)(ws + OFF_PROJ);
  float*          xconv = (float*)(ws + OFF_XC);
  float*          dtp   = (float*)(ws + OFF_DTP);
  float*          dAb   = (float*)(ws + OFF_DA);
  float*          ybuf  = (float*)(ws + OFF_Y);
  unsigned short* ybf   = (unsigned short*)(ws + OFF_YBF);

  float* out       = (float*)d_out;
  float* resid_out = out + (size_t)ROWS * HS;    // second tuple element

  // 1) bf16 weight transposes (LDS-tiled, coalesced; recomputed every call)
  {
    dim3 blk(32, 8);
    dim3 g1(HS / 32, PROJ_N / 32);       // W_in: 2048 x 10304
    k_convert_wT<<<g1, blk, 0, stream>>>(W_in, WtIn, HS, PROJ_N);
    dim3 g2(INTER / 32, HS / 32);        // W_out: 4096 x 2048
    k_convert_wT<<<g2, blk, 0, stream>>>(W_out, WtOut, INTER, HS);
  }

  // 2) residual + RMSNorm
  k_resid_norm<<<ROWS, 256, 0, stream>>>(hs, res, norm_w, resid_out, xbf);

  // 3) GEMM1: proj = x @ W_in   (M=4096, N=10304, K=2048)
  {
    dim3 grid(PROJ_N / 64, ROWS / 128);
    k_gemm_bf16<<<grid, 128, 0, stream>>>(xbf, HS, WtIn, HS, proj, PROJ_N, HS);
  }

  // 4) depthwise causal conv + SiLU
  {
    long long t = (long long)ROWS * CONV_DIM;
    k_conv_silu<<<(unsigned)((t + 255) / 256), 256, 0, stream>>>(proj, conv_w, conv_b, xconv);
  }

  // 5) dt / dA
  k_dt<<<(ROWS * HEADS + 255) / 256, 256, 0, stream>>>(proj, dt_bias, A_log, dtp, dAb);

  // 6) selective scan
  k_scan<<<BATCH * HEADS, 256, 0, stream>>>(xconv, dtp, dAb, Dp, ybuf);

  // 7) gated RMSNorm -> bf16
  k_gate_norm<<<ROWS, 256, 0, stream>>>(ybuf, proj, gate_w, ybf);

  // 8) GEMM2: out = y @ W_out   (M=4096, N=2048, K=4096)
  {
    dim3 grid(HS / 64, ROWS / 128);
    k_gemm_bf16<<<grid, 128, 0, stream>>>(ybf, INTER, WtOut, INTER, out, HS, INTER);
  }
}
